// CryptoTransformerNetwork_49426483642371
// MI455X (gfx1250) — matrix operand order repaired and checked by reading, not compiled, not run on hardware
//
#include <hip/hip_runtime.h>
#include <math.h>

#define B_   256
#define S_   100
#define DIN_ 64
#define D_   256
#define H_   8
#define L_   6
#define DFF_ 1024
#define DK_  32
#define NT_  (B_ * S_)   // 25600 tokens

typedef __attribute__((ext_vector_type(16))) _Float16 v16h;
typedef __attribute__((ext_vector_type(8)))  _Float16 v8h;
typedef __attribute__((ext_vector_type(8)))  float    v8f;

// ---------------------------------------------------------------------------
// CDNA5 wave32 WMMA 16x16x32 f16 fragment layouts as used here (contiguous k, not the ISA 7.12.2 table order):
//  A: lane 0-15 = row M=lane, K 0..15; lane 16-31 = row M=lane-16,
//     K 16..31  -> two contiguous 16B runs in a row-major [M][32] tile.
//  B: lane holds column N=lane%16; lanes 0-15 cover K 0..15, lanes 16-31 cover
//     K 16..31 -> one contiguous 32B run in a K-transposed [N][32] tile.
//  C/D: VGPR i, lane l -> row = i + 8*(l>=16), col = l%16.
// ---------------------------------------------------------------------------

__device__ __forceinline__ v16h make_v16(v8h lo, v8h hi) {
  return __builtin_shufflevector(lo, hi, 0, 1, 2, 3, 4, 5, 6, 7,
                                 8, 9, 10, 11, 12, 13, 14, 15);
}

// A fragment from row-major f16 LDS tile with row stride `ld` halves.
__device__ __forceinline__ v16h fragA_lds(const _Float16* tile, int ld,
                                          int k0, int lane) {
  const _Float16* row = tile + (lane & 15) * ld + k0 + (lane >> 4) * 16;
  v8h lo = *(const v8h*)row;
  v8h hi = *(const v8h*)(row + 8);
  return make_v16(lo, hi);
}

// ---------------------------------------------------------------------------
// Tiled GEMM: C[M x N] = epi(A[M x K] @ W[K x N] + bias (+ resid))
// 256 threads / 8 waves; block tile 64x128, K-slab 32 staged in LDS (f32->f16)
// through a register-prefetch pipeline (next slab's global loads overlap the
// current slab's WMMA work). Wave (wm = wave>>2, wn = wave&3) owns a 32x32
// register tile -> 4 WMMA per k-step. EPI: 0 none, 1 relu, 2 pos-encoding.
// ---------------------------------------------------------------------------
template <int EPI>
__global__ __launch_bounds__(256) void wmma_gemm(
    const float* __restrict__ A, const float* __restrict__ W,
    const float* __restrict__ bias, const float* __restrict__ resid,
    float* __restrict__ C, int M, int N, int K) {
  __shared__ __attribute__((aligned(32))) _Float16 As[64 * 32];   // [m][k]
  __shared__ __attribute__((aligned(32))) _Float16 Bs[128 * 32];  // [n][k]

  const int tid  = threadIdx.x;
  const int lane = tid & 31;
  const int wave = tid >> 5;
  const int wm   = wave >> 2;  // 0..1
  const int wn   = wave & 3;   // 0..3
  const int m0   = blockIdx.y * 64;
  const int n0   = blockIdx.x * 128;

  // cooperative staging coordinates (coalesced 8-float runs)
  const int ar  = tid >> 2;         // 0..63   A row
  const int ac  = (tid & 3) * 8;    // 0..24   A col-base
  const int bkr = tid >> 4;         // 0..15   B k-row (two passes: +0, +16)
  const int bn8 = (tid & 15) * 8;   // 0..120  B n-base

  float fa[8], fb[2][8];

  auto load_slab = [&](int k0) {
    const float* sa = A + (size_t)(m0 + ar) * K + k0 + ac;
#pragma unroll
    for (int j = 0; j < 8; ++j) fa[j] = sa[j];
#pragma unroll
    for (int p = 0; p < 2; ++p) {
      const float* sb = W + (size_t)(k0 + bkr + p * 16) * N + n0 + bn8;
#pragma unroll
      for (int j = 0; j < 8; ++j) fb[p][j] = sb[j];
    }
  };
  auto store_slab = [&]() {
    _Float16* da = As + ar * 32 + ac;
#pragma unroll
    for (int j = 0; j < 8; ++j) da[j] = (_Float16)fa[j];
#pragma unroll
    for (int p = 0; p < 2; ++p)
#pragma unroll
      for (int j = 0; j < 8; ++j)
        Bs[(bn8 + j) * 32 + bkr + p * 16] = (_Float16)fb[p][j];
  };

  v8f acc[2][2] = {};

  load_slab(0);
  for (int k0 = 0; k0 < K; k0 += 32) {
    __syncthreads();  // previous iteration's readers are done with LDS
    store_slab();
    __syncthreads();  // slab published
    if (k0 + 32 < K) load_slab(k0 + 32);  // overlap next slab with WMMAs
    if (k0 + 64 < K) {                    // and prefetch two slabs ahead
      __builtin_prefetch(A + (size_t)(m0 + ar) * K + (k0 + 64) + ac, 0, 3);
      __builtin_prefetch(W + (size_t)(k0 + 64 + bkr) * N + n0 + bn8, 0, 3);
    }

    const v16h a0 = fragA_lds(As + (wm * 32) * 32, 32, 0, lane);
    const v16h a1 = fragA_lds(As + (wm * 32 + 16) * 32, 32, 0, lane);
    const _Float16* bp = Bs + (wn * 32 + (lane & 15)) * 32 + (lane >> 4) * 16;
    const v16h b0 = *(const v16h*)bp;
    const v16h b1 = *(const v16h*)(bp + 16 * 32);
    acc[0][0] = __builtin_amdgcn_wmma_f32_16x16x32_f16(false, a0, false, b0,
                                                       (short)0, acc[0][0], false, false);
    acc[0][1] = __builtin_amdgcn_wmma_f32_16x16x32_f16(false, a0, false, b1,
                                                       (short)0, acc[0][1], false, false);
    acc[1][0] = __builtin_amdgcn_wmma_f32_16x16x32_f16(false, a1, false, b0,
                                                       (short)0, acc[1][0], false, false);
    acc[1][1] = __builtin_amdgcn_wmma_f32_16x16x32_f16(false, a1, false, b1,
                                                       (short)0, acc[1][1], false, false);
  }

#pragma unroll
  for (int sm = 0; sm < 2; ++sm)
#pragma unroll
    for (int sn = 0; sn < 2; ++sn) {
      const v8f acv = acc[sm][sn];
      const int col = n0 + wn * 32 + sn * 16 + (lane & 15);
      const float bv = bias[col];
#pragma unroll
      for (int i = 0; i < 8; ++i) {
        const int row = m0 + wm * 32 + sm * 16 + ((lane >> 4) << 3) + i;
        float v = acv[i] + bv;
        if (resid) v += resid[(size_t)row * N + col];
        if (EPI == 1) v = fmaxf(v, 0.f);
        if (EPI == 2) {  // sinusoidal positional encoding
          const int s  = row % S_;
          const int de = col & ~1;
          const float ang =
              (float)s * __expf(-(float)de * (9.2103403719761836f / (float)D_));
          v += (col & 1) ? __cosf(ang) : __sinf(ang);
        }
        C[(size_t)row * N + col] = v;
      }
    }
}

// ---------------------------------------------------------------------------
// Fused attention: one block per (batch, head). S=100 padded to 112 (7 tiles),
// DK=32. K/V staged in LDS as f16 once; per 16-row query tile:
//   scores (WMMA, 7 n-tiles across waves) -> masked f32 softmax ->
//   P (f16, K padded to 128) @ V (WMMA) -> O.
// ---------------------------------------------------------------------------
__global__ __launch_bounds__(256) void attn_kernel(
    const float* __restrict__ Q, const float* __restrict__ Kv,
    const float* __restrict__ V, float* __restrict__ O) {
  __shared__ __attribute__((aligned(32))) _Float16 ks[112 * 32];
  __shared__ __attribute__((aligned(32))) _Float16 vs[112 * 32];
  __shared__ __attribute__((aligned(32))) _Float16 qs[16 * 32];
  __shared__ __attribute__((aligned(32))) float    sc[16 * 112];
  __shared__ __attribute__((aligned(32))) _Float16 ps[16 * 128];

  const int tid  = threadIdx.x;
  const int lane = tid & 31;
  const int wave = tid >> 5;
  const int b = blockIdx.x / H_;
  const int h = blockIdx.x % H_;
  const size_t base = (size_t)b * S_ * D_ + (size_t)h * DK_;
  const float inv_scale = 0.17677669529663687f;  // 1/sqrt(DK)

  for (int idx = tid; idx < 112 * 32; idx += 256) {
    const int s = idx >> 5, dk = idx & 31;
    const bool in = (s < S_);
    ks[idx] = (_Float16)(in ? Kv[base + (size_t)s * D_ + dk] : 0.f);
    vs[idx] = (_Float16)(in ? V [base + (size_t)s * D_ + dk] : 0.f);
  }
  __syncthreads();

  for (int mt = 0; mt < 7; ++mt) {
    const int m0 = mt * 16;
    for (int idx = tid; idx < 16 * 32; idx += 256) {
      const int s = m0 + (idx >> 5), dk = idx & 31;
      qs[idx] = (_Float16)((s < S_) ? Q[base + (size_t)s * D_ + dk] : 0.f);
    }
    __syncthreads();

    // scores = q @ k^T / sqrt(dk)
    if (wave < 7) {
      const int n0 = wave * 16;
      const v16h a = fragA_lds(qs, 32, 0, lane);
      const v16h bf =
          *(const v16h*)(ks + (n0 + (lane & 15)) * 32 + (lane >> 4) * 16);
      v8f acc = {};
      acc = __builtin_amdgcn_wmma_f32_16x16x32_f16(false, a, false, bf,
                                                   (short)0, acc, false, false);
      const int col = n0 + (lane & 15);
#pragma unroll
      for (int i = 0; i < 8; ++i) {
        const int r = ((lane >> 4) << 3) + i;
        sc[r * 112 + col] = (col < S_) ? acc[i] * inv_scale : -1e30f;
      }
    }
    __syncthreads();

    // row softmax (f32), emit f16 probabilities, zero K-pad 112..127
    if (tid < 16) {
      float mx = -1e30f;
      for (int c = 0; c < 112; ++c) mx = fmaxf(mx, sc[tid * 112 + c]);
      float sum = 0.f;
      for (int c = 0; c < 112; ++c) {
        const float e = __expf(sc[tid * 112 + c] - mx);
        sum += e;
        ps[tid * 128 + c] = (_Float16)e;
      }
      const float inv = 1.f / sum;
      for (int c = 0; c < 112; ++c)
        ps[tid * 128 + c] = (_Float16)((float)ps[tid * 128 + c] * inv);
      for (int c = 112; c < 128; ++c) ps[tid * 128 + c] = (_Float16)0.f;
    }
    __syncthreads();

    // o = p @ v  (K = 128 padded -> 4 WMMA, N = 32 -> 2 tiles)
    if (wave < 2) {
      const int n0 = wave * 16;
      v8f acc = {};
      for (int k0 = 0; k0 < 128; k0 += 32) {
        const v16h a = fragA_lds(ps, 128, k0, lane);
        const int c = n0 + (lane & 15);
        const int kb = k0 + ((lane >> 4) << 4);
        v16h bf;
#pragma unroll
        for (int j = 0; j < 16; ++j) bf[j] = vs[(kb + j) * 32 + c];
        acc = __builtin_amdgcn_wmma_f32_16x16x32_f16(false, a, false, bf,
                                                     (short)0, acc, false, false);
      }
      const int c = n0 + (lane & 15);
#pragma unroll
      for (int i = 0; i < 8; ++i) {
        const int s = m0 + ((lane >> 4) << 3) + i;
        if (s < S_) O[base + (size_t)s * D_ + c] = acc[i];
      }
    }
    __syncthreads();
  }
}

// ---------------------------------------------------------------------------
// LayerNorm over D=256: one wave per row, 8 elems/lane, shfl_xor reduction.
// ---------------------------------------------------------------------------
__global__ void ln_kernel(const float* __restrict__ X, const float* __restrict__ g,
                          const float* __restrict__ be, float* __restrict__ Y) {
  const int row  = blockIdx.x;
  const int lane = threadIdx.x;
  const float* x = X + (size_t)row * D_;
  float v[8];
  float s = 0.f, s2 = 0.f;
#pragma unroll
  for (int j = 0; j < 8; ++j) { v[j] = x[lane * 8 + j]; s += v[j]; s2 += v[j] * v[j]; }
#pragma unroll
  for (int off = 16; off; off >>= 1) {
    s  += __shfl_xor(s,  off, 32);
    s2 += __shfl_xor(s2, off, 32);
  }
  const float mean = s * (1.f / D_);
  const float var  = s2 * (1.f / D_) - mean * mean;
  const float rstd = rsqrtf(var + 1e-5f);
  float* y = Y + (size_t)row * D_;
#pragma unroll
  for (int j = 0; j < 8; ++j) {
    const int c = lane * 8 + j;
    y[c] = (v[j] - mean) * rstd * g[c] + be[c];
  }
}

// Final LN, last token only (the heads only consume h[:, -1]).
__global__ void lastln_kernel(const float* __restrict__ X, const float* __restrict__ g,
                              const float* __restrict__ be, float* __restrict__ hf,
                              float* __restrict__ outhf) {
  const int b = blockIdx.x;
  const int lane = threadIdx.x;
  const float* x = X + ((size_t)b * S_ + (S_ - 1)) * D_;
  float v[8];
  float s = 0.f, s2 = 0.f;
#pragma unroll
  for (int j = 0; j < 8; ++j) { v[j] = x[lane * 8 + j]; s += v[j]; s2 += v[j] * v[j]; }
#pragma unroll
  for (int off = 16; off; off >>= 1) {
    s  += __shfl_xor(s,  off, 32);
    s2 += __shfl_xor(s2, off, 32);
  }
  const float mean = s * (1.f / D_);
  const float var  = s2 * (1.f / D_) - mean * mean;
  const float rstd = rsqrtf(var + 1e-5f);
#pragma unroll
  for (int j = 0; j < 8; ++j) {
    const int c = lane * 8 + j;
    const float o = (v[j] - mean) * rstd * g[c] + be[c];
    hf[(size_t)b * D_ + c]    = o;
    outhf[(size_t)b * D_ + c] = o;
  }
}

// ---------------------------------------------------------------------------
// Output heads (tiny MLPs) — one block per batch row, plain VALU.
// d_out layout: action[B], regime[B*3], confidence[B], volatility[B], hf[B*256]
// ---------------------------------------------------------------------------
__global__ void heads_kernel(const float* __restrict__ hf,
    const float* Wr1, const float* br1, const float* Wr2, const float* br2,
    const float* Wa1, const float* ba1, const float* Wa2, const float* ba2,
    const float* Wa3, const float* ba3,
    const float* Wc1, const float* bc1, const float* Wc2, const float* bc2,
    const float* Wv1, const float* bv1, const float* Wv2, const float* bv2,
    float* __restrict__ out) {
  __shared__ float x[256], r1[128], a1[128], a2[64], c1[64], w1[64];
  const int b = blockIdx.x, t = threadIdx.x;
  for (int i = t; i < 256; i += 128) x[i] = hf[(size_t)b * 256 + i];
  __syncthreads();
  if (t < 128) {
    float sr = br1[t], sa = ba1[t];
    for (int i = 0; i < 256; ++i) {
      sr += x[i] * Wr1[i * 128 + t];
      sa += x[i] * Wa1[i * 128 + t];
    }
    r1[t] = fmaxf(sr, 0.f);
    a1[t] = fmaxf(sa, 0.f);
  }
  __syncthreads();
  if (t < 64) {
    float s2 = ba2[t];
    for (int i = 0; i < 128; ++i) s2 += a1[i] * Wa2[i * 64 + t];
    a2[t] = fmaxf(s2, 0.f);
    float sc = bc1[t], sv = bv1[t];
    for (int i = 0; i < 256; ++i) {
      sc += x[i] * Wc1[i * 64 + t];
      sv += x[i] * Wv1[i * 64 + t];
    }
    c1[t] = fmaxf(sc, 0.f);
    w1[t] = fmaxf(sv, 0.f);
  }
  __syncthreads();
  if (t == 0) {
    float l0 = br2[0], l1 = br2[1], l2 = br2[2];
    for (int i = 0; i < 128; ++i) {
      l0 += r1[i] * Wr2[i * 3 + 0];
      l1 += r1[i] * Wr2[i * 3 + 1];
      l2 += r1[i] * Wr2[i * 3 + 2];
    }
    const float mx = fmaxf(l0, fmaxf(l1, l2));
    const float e0 = __expf(l0 - mx), e1 = __expf(l1 - mx), e2 = __expf(l2 - mx);
    const float inv = 1.f / (e0 + e1 + e2);
    out[B_ + b * 3 + 0] = e0 * inv;
    out[B_ + b * 3 + 1] = e1 * inv;
    out[B_ + b * 3 + 2] = e2 * inv;
    float sa3 = ba3[0];
    for (int i = 0; i < 64; ++i) sa3 += a2[i] * Wa3[i];
    out[b] = tanhf(sa3);
    float sc2 = bc2[0];
    for (int i = 0; i < 64; ++i) sc2 += c1[i] * Wc2[i];
    out[4 * B_ + b] = 1.f / (1.f + __expf(-sc2));
    float sv2 = bv2[0];
    for (int i = 0; i < 64; ++i) sv2 += w1[i] * Wv2[i];
    out[5 * B_ + b] = fmaxf(sv2, 0.f);
  }
}

// ---------------------------------------------------------------------------
extern "C" void kernel_launch(void* const* d_in, const int* in_sizes, int n_in,
                              void* d_out, int out_size, void* d_ws, size_t ws_size,
                              hipStream_t stream) {
  (void)in_sizes; (void)n_in; (void)out_size; (void)ws_size;
  const float* x     = (const float*)d_in[0];
  const float* W_in  = (const float*)d_in[1];
  const float* b_in  = (const float*)d_in[2];
  const float* Wq    = (const float*)d_in[3];
  const float* bq    = (const float*)d_in[4];
  const float* Wk    = (const float*)d_in[5];
  const float* bk    = (const float*)d_in[6];
  const float* Wv    = (const float*)d_in[7];
  const float* bv    = (const float*)d_in[8];
  const float* Wo    = (const float*)d_in[9];
  const float* bo    = (const float*)d_in[10];
  const float* g1    = (const float*)d_in[11];
  const float* be1   = (const float*)d_in[12];
  const float* W1    = (const float*)d_in[13];
  const float* b1    = (const float*)d_in[14];
  const float* W2    = (const float*)d_in[15];
  const float* b2    = (const float*)d_in[16];
  const float* g2    = (const float*)d_in[17];
  const float* be2   = (const float*)d_in[18];
  const float* lnf_g = (const float*)d_in[19];
  const float* lnf_b = (const float*)d_in[20];

  float* out = (float*)d_out;
  float* ws  = (float*)d_ws;

  const size_t HD = (size_t)NT_ * D_;
  float* h   = ws;            // [NT, D]   master activations
  float* qb  = h  + HD;       // [NT, D]   Q   (reused as pre-LN tmp t1)
  float* kb  = qb + HD;       // [NT, D]   K
  float* vb  = kb + HD;       // [NT, D]   V
  float* fb  = vb + HD;       // [NT, DFF] FFN act (also attention output buf)
  float* hfb = fb + (size_t)NT_ * DFF_;  // [B, D]
  float* ao  = fb;            // attention output aliases FFN buffer
  float* t1  = qb;            // pre-LN tmp aliases Q buffer

  const dim3 blk(256);
  const dim3 gridD(D_ / 128, NT_ / 64);     // N=256
  const dim3 gridF(DFF_ / 128, NT_ / 64);   // N=1024

  // h = x @ W_in + b_in + PE
  wmma_gemm<2><<<gridD, blk, 0, stream>>>(x, W_in, b_in, nullptr, h, NT_, D_, DIN_);

  for (int l = 0; l < L_; ++l) {
    const float* Wq_l = Wq + (size_t)l * D_ * D_;
    const float* Wk_l = Wk + (size_t)l * D_ * D_;
    const float* Wv_l = Wv + (size_t)l * D_ * D_;
    const float* Wo_l = Wo + (size_t)l * D_ * D_;
    const float* W1_l = W1 + (size_t)l * D_ * DFF_;
    const float* W2_l = W2 + (size_t)l * DFF_ * D_;

    wmma_gemm<0><<<gridD, blk, 0, stream>>>(h, Wq_l, bq + l * D_, nullptr, qb, NT_, D_, D_);
    wmma_gemm<0><<<gridD, blk, 0, stream>>>(h, Wk_l, bk + l * D_, nullptr, kb, NT_, D_, D_);
    wmma_gemm<0><<<gridD, blk, 0, stream>>>(h, Wv_l, bv + l * D_, nullptr, vb, NT_, D_, D_);

    attn_kernel<<<dim3(B_ * H_), dim3(256), 0, stream>>>(qb, kb, vb, ao);

    // t1 = h + ao @ Wo + bo ; h = LN(t1)
    wmma_gemm<0><<<gridD, blk, 0, stream>>>(ao, Wo_l, bo + l * D_, h, t1, NT_, D_, D_);
    ln_kernel<<<dim3(NT_), dim3(32), 0, stream>>>(t1, g1 + l * D_, be1 + l * D_, h);

    // f = relu(h @ W1 + b1) ; t1 = h + f @ W2 + b2 ; h = LN(t1)
    wmma_gemm<1><<<gridF, blk, 0, stream>>>(h, W1_l, b1 + l * DFF_, nullptr, fb, NT_, DFF_, D_);
    wmma_gemm<0><<<gridD, blk, 0, stream>>>(fb, W2_l, b2 + l * D_, h, t1, NT_, D_, DFF_);
    ln_kernel<<<dim3(NT_), dim3(32), 0, stream>>>(t1, g2 + l * D_, be2 + l * D_, h);
  }

  // hf = LN(h[:, -1]) -> also written straight into d_out's hf slot
  lastln_kernel<<<dim3(B_), dim3(32), 0, stream>>>(h, lnf_g, lnf_b, hfb, out + 6 * B_);

  heads_kernel<<<dim3(B_), dim3(128), 0, stream>>>(hfb,
      (const float*)d_in[21], (const float*)d_in[22], (const float*)d_in[23], (const float*)d_in[24],
      (const float*)d_in[25], (const float*)d_in[26], (const float*)d_in[27], (const float*)d_in[28],
      (const float*)d_in[29], (const float*)d_in[30],
      (const float*)d_in[31], (const float*)d_in[32], (const float*)d_in[33], (const float*)d_in[34],
      (const float*)d_in[35], (const float*)d_in[36], (const float*)d_in[37], (const float*)d_in[38],
      out);
}
